// RecommendationModel_2473901163080
// MI455X (gfx1250) — matrix operand order repaired and checked by reading, not compiled, not run on hardware
//
#include <hip/hip_runtime.h>
#include <math.h>

// ---------------- CDNA5 WMMA plumbing ----------------
typedef __attribute__((ext_vector_type(16))) _Float16 v16h;
typedef __attribute__((ext_vector_type(8)))  float    v8f;

__device__ __forceinline__ v8f wmma_f16(v16h a, v16h b, v8f c) {
  // v_wmma_f32_16x16x32_f16 : D = A(16x32 f16) x B(32x16 f16) + C(16x16 f32)
  return __builtin_amdgcn_wmma_f32_16x16x32_f16(false, a, false, b, (short)0, c, false, false);
}

__device__ __forceinline__ int imin_i(int a, int b) { return a < b ? a : b; }

__device__ __forceinline__ float gelu_tanh(float x) {
  float x3 = x * x * x;
  return 0.5f * x * (1.0f + tanhf(0.79788456080286535588f * (x + 0.044715f * x3)));
}

// ---------------- constants (match reference) ----------------
#define Hdim    128
#define HEADS   4
#define DHd     32
#define NLAYERS 3
#define N_USER  10000
#define N_ITEM  10000
#define N_TASTE 10000
#define N_INT   10000
#define N_IMG   10000
#define N_ING   5000
#define E_PART  80000
#define E_CONT  60000
#define E_AST   10000
#define E_BUYS  100000

// =====================================================================
// LDS-tiled 128x128 GEMM:  C[M,128] = epi(A[M,128] @ W[128,128] + b)
// Block = 256 threads = 8 waves, computes a 64x128 output tile.
// A tile staged in LDS as f16 (16 KB); each wave owns one 16-col stripe,
// loads its B fragment once per K-step and reuses it across 4 M sub-tiles
// -> 16 WMMAs per wave, 4x less W traffic than a 16-row-per-block scheme.
// epi: 0 = none, 1 = relu, 2 = gelu-on-A + sigmoid-skip blend with XSin.
// =====================================================================
__global__ __launch_bounds__(256) void gemm128_tiled(
    const float* __restrict__ A, const float* __restrict__ W,
    const float* __restrict__ bias, float* __restrict__ C,
    const float* __restrict__ XSin, const float* __restrict__ skipPtr,
    int M, int epi)
{
  __shared__ _Float16 As[64 * Hdim];

  const int row0 = blockIdx.x * 64;

  // ---- cooperative stage of the 64x128 A tile (fp32 -> f16, fused gelu) ----
  for (int idx = threadIdx.x; idx < 64 * Hdim; idx += 256) {
    int r = idx >> 7, c = idx & 127;
    const float* ap = &A[(size_t)imin_i(row0 + r, M - 1) * Hdim + c];
    __builtin_prefetch(ap + 64 * Hdim, 0, 0);   // global_prefetch_b8 toward next block's tile
    float v = *ap;
    if (epi == 2) v = gelu_tanh(v);
    As[idx] = (_Float16)v;
  }
  __syncthreads();

  const int lane = threadIdx.x & 31;
  const int wave = threadIdx.x >> 5;
  const int col0 = wave * 16;
  const int half = lane >> 4;                   // 0: lanes 0-15, 1: lanes 16-31
  const int nB   = col0 + (lane & 15);

  v8f acc[4] = {};
#pragma unroll
  for (int kk = 0; kk < 128; kk += 32) {
    // B fragment: 32x16 f16. lanes 0-15 hold K=kk+0..7,16..23, lanes 16-31 K=kk+8..15,24..31.
    v16h b;
    const int kb = kk + half * 8;
#pragma unroll
    for (int i = 0; i < 16; ++i)
      b[i] = (_Float16)W[(size_t)(kb + i + (i >= 8 ? 8 : 0)) * Hdim + nB];
    // reuse B across 4 M sub-tiles; A fragments come from LDS
#pragma unroll
    for (int mt = 0; mt < 4; ++mt) {
      const _Float16* arow = &As[(mt * 16 + (lane & 15)) * Hdim + kk + half * 8];
      v16h a;
#pragma unroll
      for (int i = 0; i < 8; ++i) {
        a[i]     = arow[i];
        a[i + 8] = arow[16 + i];
      }
      acc[mt] = wmma_f16(a, b, acc[mt]);
    }
  }

  // ---- epilogue. C layout: VGPR r -> M = mt*16 + half*8 + r, N = nB ----
  const float bn = bias[nB];
  const float aS = (epi == 2) ? 1.0f / (1.0f + __expf(-skipPtr[0])) : 0.0f;
#pragma unroll
  for (int mt = 0; mt < 4; ++mt) {
    const int mbase = row0 + mt * 16 + half * 8;
#pragma unroll
    for (int r = 0; r < 8; ++r) {
      int m = mbase + r;
      if (m < M) {
        size_t off = (size_t)m * Hdim + nB;
        float v = acc[mt][r] + bn;
        if (epi == 1)      v = fmaxf(v, 0.0f);
        else if (epi == 2) v = aS * v + (1.0f - aS) * XSin[off];
        C[off] = v;
      }
    }
  }
}

// =====================================================================
// Per-edge relation-transformed keys + attention scores.
// Block = 16 edges. Wave w: head = w>>1, f-tile = (w&1)*16.
//   ke[e,h,f] = sum_d K[src,h,d]*arel[h,d,f]  (one WMMA per wave, K=DH=32)
// then 64 threads do the 32-long dot with Q[dst] out of LDS.
// =====================================================================
__global__ __launch_bounds__(256) void edge_score_wmma(
    const int* __restrict__ src, const int* __restrict__ dst, int E,
    const float* __restrict__ Ksrc, const float* __restrict__ Qdst,
    const float* __restrict__ arel, const float* __restrict__ p,
    float* __restrict__ score)
{
  __shared__ float ke_s[16 * Hdim];
  const int lane  = threadIdx.x & 31;
  const int wave  = threadIdx.x >> 5;
  const int e0    = blockIdx.x * 16;
  const int h     = wave >> 1;
  const int ftile = (wave & 1) * 16;
  const int half  = lane >> 4;

  const int eA = imin_i(e0 + (lane & 15), E - 1);
  const int sA = src[eA];
  v16h a, b;
  const float* krow = Ksrc + (size_t)sA * Hdim + h * DHd + half * 8;
#pragma unroll
  for (int i = 0; i < 8; ++i) {
    a[i]     = (_Float16)krow[i];
    a[i + 8] = (_Float16)krow[16 + i];
  }
  const int nB = ftile + (lane & 15);
  const int kb = half * 8;
#pragma unroll
  for (int i = 0; i < 16; ++i)
    b[i] = (_Float16)arel[(size_t)(h * DHd + kb + i + (i >= 8 ? 8 : 0)) * DHd + nB];
  v8f acc = {};
  acc = wmma_f16(a, b, acc);

  const int n = ftile + (lane & 15);
#pragma unroll
  for (int r = 0; r < 8; ++r)
    ke_s[(half * 8 + r) * Hdim + h * DHd + n] = acc[r];
  __syncthreads();

  const int t = threadIdx.x;
  if (t < 64) {
    int e = t >> 2, hh = t & 3;
    int ge = e0 + e;
    if (ge < E) {
      int d = dst[ge];
      const float* q  = Qdst + (size_t)d * Hdim + hh * DHd;
      const float* kk = &ke_s[e * Hdim + hh * DHd];
      float s = 0.0f;
#pragma unroll
      for (int f = 0; f < DHd; ++f) s += kk[f] * q[f];
      score[(size_t)ge * HEADS + hh] = s * p[hh] * 0.17677669529663687f; // /sqrt(32)
    }
  }
}

// =====================================================================
// Recompute relation-transformed messages via WMMA, apply softmax weight,
// atomically aggregate into agg[dst].
// =====================================================================
__global__ __launch_bounds__(256) void edge_agg_wmma(
    const int* __restrict__ src, const int* __restrict__ dst, int E,
    const float* __restrict__ Vsrc, const float* __restrict__ mrel,
    const float* __restrict__ score, const float* __restrict__ mstat,
    const float* __restrict__ zstat, float* __restrict__ agg)
{
  __shared__ float ve_s[16 * Hdim];
  const int lane  = threadIdx.x & 31;
  const int wave  = threadIdx.x >> 5;
  const int e0    = blockIdx.x * 16;
  const int h     = wave >> 1;
  const int ftile = (wave & 1) * 16;
  const int half  = lane >> 4;

  const int eA = imin_i(e0 + (lane & 15), E - 1);
  const int sA = src[eA];
  v16h a, b;
  const float* vrow = Vsrc + (size_t)sA * Hdim + h * DHd + half * 8;
#pragma unroll
  for (int i = 0; i < 8; ++i) {
    a[i]     = (_Float16)vrow[i];
    a[i + 8] = (_Float16)vrow[16 + i];
  }
  const int nB = ftile + (lane & 15);
  const int kb = half * 8;
#pragma unroll
  for (int i = 0; i < 16; ++i)
    b[i] = (_Float16)mrel[(size_t)(h * DHd + kb + i + (i >= 8 ? 8 : 0)) * DHd + nB];
  v8f acc = {};
  acc = wmma_f16(a, b, acc);

  const int n = ftile + (lane & 15);
#pragma unroll
  for (int r = 0; r < 8; ++r)
    ve_s[(half * 8 + r) * Hdim + h * DHd + n] = acc[r];
  __syncthreads();

  for (int idx = threadIdx.x; idx < 16 * Hdim; idx += 256) {
    int e = idx >> 7, c = idx & 127, hh = c >> 5;
    int ge = e0 + e;
    if (ge < E) {
      int d = dst[ge];
      float m = mstat[(size_t)d * HEADS + hh];
      float z = zstat[(size_t)d * HEADS + hh];
      float att = __expf(score[(size_t)ge * HEADS + hh] - m) / (z + 1e-16f);
      atomicAdd(&agg[(size_t)d * Hdim + c], att * ve_s[idx]);
    }
  }
}

// ---------------- segment softmax statistics ----------------
__global__ void seg_max_kernel(const int* __restrict__ dst,
                               const float* __restrict__ score, int E,
                               unsigned* __restrict__ menc)
{
  int i = blockIdx.x * blockDim.x + threadIdx.x;
  if (i < E * HEADS) {
    int e = i >> 2, h = i & 3;
    unsigned u = __float_as_uint(score[i]);
    u = (u & 0x80000000u) ? ~u : (u | 0x80000000u);   // order-preserving map
    atomicMax(&menc[(size_t)dst[e] * HEADS + h], u);
  }
}

__global__ void decode_max_kernel(unsigned* __restrict__ m, int n)
{
  int i = blockIdx.x * blockDim.x + threadIdx.x;
  if (i < n) {
    unsigned u = m[i];
    m[i] = (u & 0x80000000u) ? (u & 0x7FFFFFFFu) : ~u;
  }
}

__global__ void seg_expsum_kernel(const int* __restrict__ dst,
                                  const float* __restrict__ score,
                                  const float* __restrict__ m, int E,
                                  float* __restrict__ z)
{
  int i = blockIdx.x * blockDim.x + threadIdx.x;
  if (i < E * HEADS) {
    int e = i >> 2, h = i & 3;
    int d = dst[e];
    atomicAdd(&z[(size_t)d * HEADS + h],
              __expf(score[i] - m[(size_t)d * HEADS + h]));
  }
}

// ---------------- ingredient -> taste mean aggregation ----------------
__global__ void ingagg_kernel(const int* __restrict__ src, const int* __restrict__ dst,
                              int E, const float* __restrict__ xing,
                              float* __restrict__ agg, float* __restrict__ cnt)
{
  int i = blockIdx.x * blockDim.x + threadIdx.x;
  if (i < E * Hdim) {
    int e = i >> 7, c = i & 127;
    int d = dst[e];
    atomicAdd(&agg[(size_t)d * Hdim + c], xing[(size_t)src[e] * Hdim + c]);
    if (c == 0) atomicAdd(&cnt[d], 1.0f);
  }
}

__global__ void divclamp_kernel(const float* __restrict__ agg,
                                const float* __restrict__ cnt,
                                float* __restrict__ outv, int N)
{
  int i = blockIdx.x * blockDim.x + threadIdx.x;
  if (i < N * Hdim) outv[i] = agg[i] / fmaxf(cnt[i >> 7], 1.0f);
}

// ---------------- LGConv ----------------
__global__ void deg_kernel(const int* __restrict__ dst, int E, float* __restrict__ deg)
{
  int i = blockIdx.x * blockDim.x + threadIdx.x;
  if (i < E) atomicAdd(&deg[dst[i]], 1.0f);
}

__global__ void lgconv_kernel(const int* __restrict__ src, const int* __restrict__ dst,
                              int E, const float* __restrict__ deg,
                              const float* __restrict__ x, float* __restrict__ y)
{
  int i = blockIdx.x * blockDim.x + threadIdx.x;
  if (i < E * Hdim) {
    int e = i >> 7, c = i & 127;
    int s = src[e], d = dst[e];
    float ds = deg[s], dd = deg[d];
    float norm = (ds > 0.f ? rsqrtf(ds) : 0.f) * (dd > 0.f ? rsqrtf(dd) : 0.f);
    atomicAdd(&y[(size_t)d * Hdim + c], norm * x[(size_t)s * Hdim + c]);
  }
}

// ---------------- nutrient projection + contrastive loss ----------------
__global__ __launch_bounds__(128) void nut_loss_kernel(
    const float* __restrict__ nutrient, const float* __restrict__ W,
    const float* __restrict__ b, const float* __restrict__ xint,
    float* __restrict__ loss, int N)
{
  __shared__ float red[3][128];
  int row = blockIdx.x, c = threadIdx.x;
  float nh = b[c];
#pragma unroll
  for (int k = 0; k < 20; ++k) nh += nutrient[(size_t)row * 20 + k] * W[(size_t)k * Hdim + c];
  float xv = xint[(size_t)row * Hdim + c];
  red[0][c] = nh * nh; red[1][c] = xv * xv; red[2][c] = nh * xv;
  __syncthreads();
  for (int s = 64; s > 0; s >>= 1) {
    if (c < s) { red[0][c] += red[0][c + s]; red[1][c] += red[1][c + s]; red[2][c] += red[2][c + s]; }
    __syncthreads();
  }
  if (c == 0) {
    float n1 = sqrtf(red[0][0]) + 1e-8f;
    float n2 = sqrtf(red[1][0]) + 1e-8f;
    atomicAdd(loss, (1.0f - red[2][0] / (n1 * n2)) / (float)N);
  }
}

// =====================================================================
// Host orchestration
// =====================================================================
extern "C" void kernel_launch(void* const* d_in, const int* in_sizes, int n_in,
                              void* d_out, int out_size, void* d_ws, size_t ws_size,
                              hipStream_t stream)
{
  const float* x_user = (const float*)d_in[0];
  const float* x_item = (const float*)d_in[1];
  const float* x_taste = (const float*)d_in[2];
  const float* x_int  = (const float*)d_in[3];
  const float* x_img  = (const float*)d_in[4];
  const float* x_ing  = (const float*)d_in[5];
  const float* nutrient = (const float*)d_in[6];
  const float* nut_W  = (const float*)d_in[7];
  const float* nut_b  = (const float*)d_in[8];
  const float* proj_W = (const float*)d_in[9];
  const float* proj_b = (const float*)d_in[10];
  const float* kqvW   = (const float*)d_in[11];
  const float* kqvB   = (const float*)d_in[12];
  const float* arelW  = (const float*)d_in[13];
  const float* mrelW  = (const float*)d_in[14];
  const float* pW     = (const float*)d_in[15];
  const float* outW   = (const float*)d_in[16];
  const float* outB   = (const float*)d_in[17];
  const float* skipW  = (const float*)d_in[18];
  const int* eiPart = (const int*)d_in[19];
  const int* eiCont = (const int*)d_in[20];
  const int* eiAT   = (const int*)d_in[21];
  const int* eiAN   = (const int*)d_in[22];
  const int* eiAM   = (const int*)d_in[23];
  const int* eiB    = (const int*)d_in[24];

  float* out = (float*)d_out;
  float* ws  = (float*)d_ws;

  // ---- workspace layout (floats) ----
  size_t o = 0;
  auto alloc = [&](size_t n) { float* p = ws + o; o += n; return p; };
  float* xs[5];
  xs[0] = alloc((size_t)N_USER * Hdim);
  xs[1] = alloc((size_t)N_ITEM * Hdim);
  xs[2] = alloc((size_t)N_TASTE * Hdim);
  xs[3] = alloc((size_t)N_INT * Hdim);
  xs[4] = alloc((size_t)N_IMG * Hdim);
  float* xs_ing = alloc((size_t)N_ING * Hdim);
  float* Kb[5], *Vb[5], *Qb[2];
  for (int i = 0; i < 5; ++i) Kb[i] = alloc((size_t)10000 * Hdim);
  for (int i = 0; i < 5; ++i) Vb[i] = alloc((size_t)10000 * Hdim);
  Qb[0] = alloc((size_t)N_USER * Hdim);
  Qb[1] = alloc((size_t)N_ITEM * Hdim);
  float* scoreBuf = alloc((size_t)(E_AST * 3 + E_BUYS * 2) * HEADS);
  float* mBuf[2]  = { alloc((size_t)N_USER * HEADS), alloc((size_t)N_ITEM * HEADS) };
  float* zBuf[2]  = { alloc((size_t)N_USER * HEADS), alloc((size_t)N_ITEM * HEADS) };
  float* aggBuf[2]= { alloc((size_t)N_USER * Hdim),  alloc((size_t)N_ITEM * Hdim) };
  float* tmpT = alloc((size_t)N_TASTE * Hdim);
  float* degT = alloc(N_TASTE);
  float* cntT = alloc(N_TASTE);
  float* aggT = alloc((size_t)N_TASTE * Hdim);
  if (ws_size < o * sizeof(float)) return;

  const int nNodes[5] = { N_USER, N_ITEM, N_TASTE, N_INT, N_IMG };

  // ---- 1. input projections (ReLU). taste projection is dead (overwritten). ----
  const float* rawX[6] = { x_user, x_item, x_taste, x_int, x_img, x_ing };
  for (int i = 0; i < 6; ++i) {
    if (i == 2) continue;
    int M = (i == 5) ? N_ING : 10000;
    float* dstp = (i == 5) ? xs_ing : xs[i];
    gemm128_tiled<<<dim3((M + 63) / 64), dim3(256), 0, stream>>>(
        rawX[i], proj_W + (size_t)i * Hdim * Hdim, proj_b + (size_t)i * Hdim,
        dstp, nullptr, nullptr, M, 1);
  }

  // ---- 2. ingredient -> taste mean aggregation ----
  hipMemsetAsync(aggT, 0, (size_t)N_TASTE * Hdim * sizeof(float), stream);
  hipMemsetAsync(cntT, 0, (size_t)N_TASTE * sizeof(float), stream);
  {
    int n = E_PART * Hdim;
    ingagg_kernel<<<(n + 255) / 256, 256, 0, stream>>>(
        eiPart, eiPart + E_PART, E_PART, xs_ing, aggT, cntT);
    int m = N_TASTE * Hdim;
    divclamp_kernel<<<(m + 255) / 256, 256, 0, stream>>>(aggT, cntT, xs[2], N_TASTE);
  }

  // ---- 3. LGConv x3 on taste graph ----
  hipMemsetAsync(degT, 0, (size_t)N_TASTE * sizeof(float), stream);
  deg_kernel<<<(E_CONT + 255) / 256, 256, 0, stream>>>(eiCont + E_CONT, E_CONT, degT);
  {
    const int* s = eiCont;
    const int* d = eiCont + E_CONT;
    int n = E_CONT * Hdim;
    hipMemsetAsync(tmpT, 0, (size_t)N_TASTE * Hdim * sizeof(float), stream);
    lgconv_kernel<<<(n + 255) / 256, 256, 0, stream>>>(s, d, E_CONT, degT, xs[2], tmpT);
    hipMemsetAsync(xs[2], 0, (size_t)N_TASTE * Hdim * sizeof(float), stream);
    lgconv_kernel<<<(n + 255) / 256, 256, 0, stream>>>(s, d, E_CONT, degT, tmpT, xs[2]);
    hipMemsetAsync(tmpT, 0, (size_t)N_TASTE * Hdim * sizeof(float), stream);
    lgconv_kernel<<<(n + 255) / 256, 256, 0, stream>>>(s, d, E_CONT, degT, xs[2], tmpT);
    hipMemcpyAsync(xs[2], tmpT, (size_t)N_TASTE * Hdim * sizeof(float),
                   hipMemcpyDeviceToDevice, stream);
  }

  // ---- 4. contrastive loss (uses raw x_intention) ----
  hipMemsetAsync(out + (size_t)5 * 10000 * Hdim, 0, sizeof(float), stream);
  nut_loss_kernel<<<N_INT, 128, 0, stream>>>(
      nutrient, nut_W, nut_b, x_int, out + (size_t)5 * 10000 * Hdim, N_INT);

  // ---- 5. HGT layers ----
  struct EM { const int* src; const int* dst; int E; int sIdx; int dIdx; size_t so; };
  const EM em[5] = {
    { eiAT,          eiAT + E_AST,  E_AST,  2, 1, 0 },
    { eiAN,          eiAN + E_AST,  E_AST,  3, 1, (size_t)E_AST * HEADS },
    { eiAM,          eiAM + E_AST,  E_AST,  4, 1, (size_t)2 * E_AST * HEADS },
    { eiB,           eiB + E_BUYS,  E_BUYS, 0, 1, (size_t)3 * E_AST * HEADS },
    { eiB + E_BUYS,  eiB,           E_BUYS, 1, 0, (size_t)3 * E_AST * HEADS + (size_t)E_BUYS * HEADS },
  };

  for (int l = 0; l < NLAYERS; ++l) {
    // K/V projections for all 5 node types; Q only for user(0)/item(1)
    for (int i = 0; i < 5; ++i) {
      int M = nNodes[i];
      const float* Wk = kqvW + ((size_t)(l * 3 + 0) * 5 + i) * Hdim * Hdim;
      const float* bk = kqvB + ((size_t)(l * 3 + 0) * 5 + i) * Hdim;
      gemm128_tiled<<<dim3((M + 63) / 64), dim3(256), 0, stream>>>(
          xs[i], Wk, bk, Kb[i], nullptr, nullptr, M, 0);
      const float* Wv = kqvW + ((size_t)(l * 3 + 2) * 5 + i) * Hdim * Hdim;
      const float* bv = kqvB + ((size_t)(l * 3 + 2) * 5 + i) * Hdim;
      gemm128_tiled<<<dim3((M + 63) / 64), dim3(256), 0, stream>>>(
          xs[i], Wv, bv, Vb[i], nullptr, nullptr, M, 0);
    }
    for (int dt = 0; dt < 2; ++dt) {
      int M = nNodes[dt];
      const float* Wq = kqvW + ((size_t)(l * 3 + 1) * 5 + dt) * Hdim * Hdim;
      const float* bq = kqvB + ((size_t)(l * 3 + 1) * 5 + dt) * Hdim;
      gemm128_tiled<<<dim3((M + 63) / 64), dim3(256), 0, stream>>>(
          xs[dt], Wq, bq, Qb[dt], nullptr, nullptr, M, 0);
    }

    // reset per-layer softmax/agg state
    for (int dt = 0; dt < 2; ++dt) {
      hipMemsetAsync(mBuf[dt], 0, (size_t)10000 * HEADS * sizeof(float), stream); // encoded -inf
      hipMemsetAsync(zBuf[dt], 0, (size_t)10000 * HEADS * sizeof(float), stream);
      hipMemsetAsync(aggBuf[dt], 0, (size_t)10000 * Hdim * sizeof(float), stream);
    }

    // scores (WMMA relation transform of keys)
    for (int e = 0; e < 5; ++e) {
      const float* ar = arelW + ((size_t)l * 5 + e) * HEADS * DHd * DHd;
      const float* pp = pW + ((size_t)l * 5 + e) * HEADS;
      edge_score_wmma<<<dim3((em[e].E + 15) / 16), dim3(256), 0, stream>>>(
          em[e].src, em[e].dst, em[e].E, Kb[em[e].sIdx], Qb[em[e].dIdx], ar, pp,
          scoreBuf + em[e].so);
    }
    // segment max
    for (int e = 0; e < 5; ++e) {
      int n = em[e].E * HEADS;
      seg_max_kernel<<<(n + 255) / 256, 256, 0, stream>>>(
          em[e].dst, scoreBuf + em[e].so, em[e].E, (unsigned*)mBuf[em[e].dIdx]);
    }
    for (int dt = 0; dt < 2; ++dt) {
      int n = 10000 * HEADS;
      decode_max_kernel<<<(n + 255) / 256, 256, 0, stream>>>((unsigned*)mBuf[dt], n);
    }
    // exp-sum
    for (int e = 0; e < 5; ++e) {
      int n = em[e].E * HEADS;
      seg_expsum_kernel<<<(n + 255) / 256, 256, 0, stream>>>(
          em[e].dst, scoreBuf + em[e].so, mBuf[em[e].dIdx], em[e].E, zBuf[em[e].dIdx]);
    }
    // weighted message aggregation (WMMA relation transform of values)
    for (int e = 0; e < 5; ++e) {
      const float* mr = mrelW + ((size_t)l * 5 + e) * HEADS * DHd * DHd;
      edge_agg_wmma<<<dim3((em[e].E + 15) / 16), dim3(256), 0, stream>>>(
          em[e].src, em[e].dst, em[e].E, Vb[em[e].sIdx], mr,
          scoreBuf + em[e].so, mBuf[em[e].dIdx], zBuf[em[e].dIdx], aggBuf[em[e].dIdx]);
    }
    // output projections + skip blend (in place) for item(1) and user(0)
    for (int dt = 0; dt < 2; ++dt) {
      int M = nNodes[dt];
      const float* Wo = outW + ((size_t)l * 5 + dt) * Hdim * Hdim;
      const float* bo = outB + ((size_t)l * 5 + dt) * Hdim;
      const float* sk = skipW + ((size_t)l * 5 + dt);
      gemm128_tiled<<<dim3((M + 63) / 64), dim3(256), 0, stream>>>(
          aggBuf[dt], Wo, bo, xs[dt], xs[dt], sk, M, 2);
    }
  }

  // ---- 6. gather outputs: user,item,taste,intention,image then loss ----
  for (int i = 0; i < 5; ++i)
    hipMemcpyAsync(out + (size_t)i * 10000 * Hdim, xs[i],
                   (size_t)10000 * Hdim * sizeof(float),
                   hipMemcpyDeviceToDevice, stream);
}
